// MHAttention_4432406250079
// MI455X (gfx1250) — matrix operand order repaired and checked by reading, not compiled, not run on hardware
//
#include <hip/hip_runtime.h>
#include <hip/hip_bf16.h>
#include <stdint.h>

typedef _Float16 v16h __attribute__((ext_vector_type(16)));
typedef _Float16 v8h  __attribute__((ext_vector_type(8)));
typedef _Float16 v4h  __attribute__((ext_vector_type(4)));
typedef float    v8f  __attribute__((ext_vector_type(8)));

#define DIM    1024
#define SEQ    2048
#define NBATCH 2
#define NHEAD  16
#define HD     64

#define WMMA_F16(a, b, c) \
  __builtin_amdgcn_wmma_f32_16x16x32_f16(false, (a), false, (b), (short)0, (c), false, false)

// ---------------------------------------------------------------------------
// CDNA5 async global->LDS copy (16B per lane), tracked by ASYNCcnt.
// GV mode: 64-bit global address in VGPR pair, LDS byte offset in VGPR.
// ---------------------------------------------------------------------------
__device__ __forceinline__ void async_copy_b128(uint32_t lds_off, const void* gsrc) {
  asm volatile("global_load_async_to_lds_b128 %0, %1, off"
               :: "v"(lds_off), "v"((uint64_t)(uintptr_t)gsrc)
               : "memory");
}
__device__ __forceinline__ uint32_t lds_addr(const void* p) {
  return (uint32_t)(uintptr_t)p;
}
#define S_WAIT_ASYNC0() asm volatile("s_wait_asynccnt 0" ::: "memory")
#define S_WAIT_ASYNC2() asm volatile("s_wait_asynccnt 2" ::: "memory")
#define S_WAIT_ASYNC4() asm volatile("s_wait_asynccnt 4" ::: "memory")

// ---------------------------------------------------------------------------
// Fragment loaders (wave32; this file's contiguous k order, same for A and B).
// A 16x32 f16 frag: lane lm = row; cols [hf*16 .. hf*16+15]
// B 32x16 f16 frag: lane lm = column (row of row-major source); K contiguous:
//   cols [hf*16 .. hf*16+15]
// ---------------------------------------------------------------------------
__device__ __forceinline__ v16h load_a16_g(const _Float16* base, int stride, int lm, int hf) {
  const _Float16* p = base + (size_t)lm * stride + hf * 16;
  v8h lo = *(const v8h*)p;
  v8h hi = *(const v8h*)(p + 8);
  return __builtin_shufflevector(lo, hi, 0, 1, 2, 3, 4, 5, 6, 7, 8, 9, 10, 11, 12, 13, 14, 15);
}
__device__ __forceinline__ v16h load_b16_g(const _Float16* base, int stride, int lm, int hf) {
  const _Float16* p = base + (size_t)lm * stride + hf * 16;
  return *(const v16h*)p;
}
// LDS variants: always two 16B loads (padded strides are not 32B multiples)
__device__ __forceinline__ v16h load_a16_lds(const _Float16* base, int stride, int lm, int hf) {
  const _Float16* p = base + lm * stride + hf * 16;
  v8h lo = *(const v8h*)p;
  v8h hi = *(const v8h*)(p + 8);
  return __builtin_shufflevector(lo, hi, 0, 1, 2, 3, 4, 5, 6, 7, 8, 9, 10, 11, 12, 13, 14, 15);
}
__device__ __forceinline__ v16h load_b16_lds(const _Float16* base, int stride, int lm, int hf,
                                             int kofs) {
  const _Float16* p = base + lm * stride + kofs + hf * 16;
  v8h lo = *(const v8h*)p;
  v8h hi = *(const v8h*)(p + 8);
  return __builtin_shufflevector(lo, hi, 0, 1, 2, 3, 4, 5, 6, 7, 8, 9, 10, 11, 12, 13, 14, 15);
}

// ---------------------------------------------------------------------------
// Kernel 0: f32 -> f16 conversion (vectorized x4)
// ---------------------------------------------------------------------------
__global__ __launch_bounds__(256) void cvt_f32_f16(const float* __restrict__ s,
                                                   _Float16* __restrict__ d, int n4) {
  int i = blockIdx.x * blockDim.x + threadIdx.x;
  if (i < n4) {
    float4 v = ((const float4*)s)[i];
    v4h h = {(_Float16)v.x, (_Float16)v.y, (_Float16)v.z, (_Float16)v.w};
    ((v4h*)d)[i] = h;
  }
}

// ---------------------------------------------------------------------------
// Straight-line DMA issue of a 64x32 A tile (stride-40 LDS rows):
// 256 chunks of 8 halfs; thread t takes chunks t and t+128 (same col, +32 rows).
// ---------------------------------------------------------------------------
__device__ __forceinline__ void issue_a_tile(const _Float16* Ag, _Float16* dst,
                                             int k, int tid) {
  const int r = tid >> 2, cc = tid & 3;
  async_copy_b128(lds_addr(dst + r * 40 + cc * 8),
                  Ag + (size_t)r * DIM + k + cc * 8);
  async_copy_b128(lds_addr(dst + (r + 32) * 40 + cc * 8),
                  Ag + (size_t)(r + 32) * DIM + k + cc * 8);
}

// ---------------------------------------------------------------------------
// Kernel 1: fused QKV projection. Block = 4 waves sharing a 64-row A tile
// (async-DMA'd to LDS, double buffered); each wave owns a 32-col N strip.
// Results scattered (with bias) into head-major Q/K and transposed Vt.
// ---------------------------------------------------------------------------
__global__ __launch_bounds__(128) void qkv_gemm(const _Float16* __restrict__ xh,
                                                const _Float16* __restrict__ wh,
                                                const float* __restrict__ bias,
                                                _Float16* __restrict__ Qh,
                                                _Float16* __restrict__ Kh,
                                                _Float16* __restrict__ Vth) {
  __shared__ _Float16 At[2][64 * 40];
  const int tid  = threadIdx.x;
  const int lane = tid & 31, wid = tid >> 5;
  const int lm = lane & 15, hf = lane >> 4;
  const int batch = blockIdx.z;
  const int mbase = blockIdx.y * 64;
  const int nbase = (blockIdx.x * 4 + wid) * 32;

  const _Float16* Ag = xh + (size_t)batch * SEQ * DIM + (size_t)mbase * DIM;
  const _Float16* B0 = wh + (size_t)nbase * DIM;
  const _Float16* B1 = wh + (size_t)(nbase + 16) * DIM;

  issue_a_tile(Ag, At[0], 0, tid);
  v8f c[4][2] = {};
  int buf = 0;
  for (int k = 0; k < DIM; k += 32, buf ^= 1) {
    if (k + 32 < DIM) {
      issue_a_tile(Ag, At[buf ^ 1], k + 32, tid);
      S_WAIT_ASYNC2();          // oldest tile's 2 DMAs complete
    } else {
      S_WAIT_ASYNC0();
    }
    __syncthreads();
    __builtin_prefetch(B0 + (size_t)lm * DIM + k + 64, 0, 1);
    v16h b0 = load_b16_g(B0 + k, DIM, lm, hf);
    v16h b1 = load_b16_g(B1 + k, DIM, lm, hf);
#pragma unroll
    for (int mi = 0; mi < 4; mi++) {
      v16h a = load_a16_lds(&At[buf][mi * 16 * 40], 40, lm, hf);
      c[mi][0] = WMMA_F16(a, b0, c[mi][0]);
      c[mi][1] = WMMA_F16(a, b1, c[mi][1]);
    }
    __syncthreads();            // everyone done reading At[buf] before reuse
  }

#pragma unroll
  for (int ni = 0; ni < 2; ni++) {
    const int obase  = nbase + ni * 16;      // uniform over wave
    const int region = obase >> 10;          // 0=q, 1=k, 2=v
    const int head   = (obase & 1023) >> 6;
    const int d      = (obase & 63) + lm;
    const float bv   = bias[obase + lm];
    const size_t bh  = (size_t)(batch * NHEAD + head);
#pragma unroll
    for (int mi = 0; mi < 4; mi++) {
#pragma unroll
      for (int r = 0; r < 8; r++) {
        const int srow = mbase + mi * 16 + r + hf * 8;
        const _Float16 h = (_Float16)(c[mi][ni][r] + bv);
        if (region == 0)      Qh [(bh * SEQ + srow) * HD + d]  = h;
        else if (region == 1) Kh [(bh * SEQ + srow) * HD + d]  = h;
        else                  Vth[(bh * HD + d) * SEQ + srow]  = h;
      }
    }
  }
}

// ---------------------------------------------------------------------------
// Kernel 2: flash attention. Block = 4 waves, each owning a 16-query tile of
// the SAME (batch, head): K (32x64, stride 72) and V^T (64x32, stride 40)
// key-block tiles are async-DMA'd into LDS double-buffered and shared by all
// 4 waves. Scores stay in WMMA accumulators (online softmax); P transposes
// C->A fragment layout through per-wave LDS staging.
// ---------------------------------------------------------------------------
__global__ __launch_bounds__(128) void attn_flash(const _Float16* __restrict__ Qh,
                                                  const _Float16* __restrict__ Kh,
                                                  const _Float16* __restrict__ Vth,
                                                  _Float16* __restrict__ ctxh) {
  __shared__ _Float16 Kt[2][32 * 72];
  __shared__ _Float16 Vts[2][64 * 40];
  __shared__ _Float16 Pl[4][16 * 40];
  const int tid  = threadIdx.x;
  const int lane = tid & 31, wid = tid >> 5;
  const int lm = lane & 15, hf = lane >> 4;
  const int qblock = blockIdx.x * 4 + wid;
  const int head = blockIdx.y, batch = blockIdx.z;
  const size_t bh = (size_t)(batch * NHEAD + head);
  const int qbase = qblock * 16;

  const _Float16* Qb = Qh + (bh * SEQ + qbase) * HD;
  const _Float16* Kb = Kh + bh * SEQ * HD;
  const _Float16* Vb = Vth + bh * HD * SEQ;
  _Float16* Pw = &Pl[wid][0];

  // straight-line issue of one key-block's K and V^T tiles (4 DMAs/thread)
  const int kr = tid >> 3, kc = tid & 7;   // K tile: 8 chunks/row
  const int vr = tid >> 2, vc = tid & 3;   // V^T tile: 4 chunks/row
  auto issue_kv = [&](int kb, int b) {
    async_copy_b128(lds_addr(&Kt[b][kr * 72 + kc * 8]),
                    Kb + (size_t)(kb + kr) * HD + kc * 8);
    async_copy_b128(lds_addr(&Kt[b][(kr + 16) * 72 + kc * 8]),
                    Kb + (size_t)(kb + kr + 16) * HD + kc * 8);
    async_copy_b128(lds_addr(&Vts[b][vr * 40 + vc * 8]),
                    Vb + (size_t)vr * SEQ + kb + vc * 8);
    async_copy_b128(lds_addr(&Vts[b][(vr + 32) * 40 + vc * 8]),
                    Vb + (size_t)(vr + 32) * SEQ + kb + vc * 8);
  };

  const v16h aQ0 = load_a16_g(Qb, HD, lm, hf);       // d-chunk 0..31
  const v16h aQ1 = load_a16_g(Qb + 32, HD, lm, hf);  // d-chunk 32..63

  float mrow[8], lrow[8];
  v8f acc[4] = {};
#pragma unroll
  for (int r = 0; r < 8; r++) { mrow[r] = -1e30f; lrow[r] = 0.0f; }

  issue_kv(0, 0);
  int buf = 0;
  for (int kb = 0; kb < SEQ; kb += 32, buf ^= 1) {
    if (kb + 32 < SEQ) {
      issue_kv(kb + 32, buf ^ 1);
      S_WAIT_ASYNC4();          // oldest tile's 4 DMAs complete
    } else {
      S_WAIT_ASYNC0();
    }
    __syncthreads();

    // ---- S = Q K^T for this 16x32 key block (two 16x16 C-frags) ----
    v8f s0 = {}, s1 = {};
    {
      v16h b00 = load_b16_lds(Kt[buf], 72, lm, hf, 0);
      v16h b01 = load_b16_lds(Kt[buf], 72, lm, hf, 32);
      v16h b10 = load_b16_lds(Kt[buf] + 16 * 72, 72, lm, hf, 0);
      v16h b11 = load_b16_lds(Kt[buf] + 16 * 72, 72, lm, hf, 32);
      s0 = WMMA_F16(aQ0, b00, s0);
      s0 = WMMA_F16(aQ1, b01, s0);
      s1 = WMMA_F16(aQ0, b10, s1);
      s1 = WMMA_F16(aQ1, b11, s1);
    }

    // ---- online softmax (rows r / r+8 live in matching lane halves) ----
#pragma unroll
    for (int r = 0; r < 8; r++) {
      float x0 = s0[r] * 0.125f;   // 1/sqrt(64)
      float x1 = s1[r] * 0.125f;
      float t = fmaxf(x0, x1);
      t = fmaxf(t, __shfl_xor(t, 1));
      t = fmaxf(t, __shfl_xor(t, 2));
      t = fmaxf(t, __shfl_xor(t, 4));
      t = fmaxf(t, __shfl_xor(t, 8));
      const float mnew = fmaxf(mrow[r], t);
      const float p0 = __expf(x0 - mnew);
      const float p1 = __expf(x1 - mnew);
      float sum = p0 + p1;
      sum += __shfl_xor(sum, 1);
      sum += __shfl_xor(sum, 2);
      sum += __shfl_xor(sum, 4);
      sum += __shfl_xor(sum, 8);
      const float corr = __expf(mrow[r] - mnew);
      lrow[r] = lrow[r] * corr + sum;
      mrow[r] = mnew;
#pragma unroll
      for (int t4 = 0; t4 < 4; t4++) acc[t4][r] *= corr;
      const int m = r + hf * 8;
      Pw[m * 40 + lm]      = (_Float16)p0;
      Pw[m * 40 + 16 + lm] = (_Float16)p1;
    }
    asm volatile("s_wait_dscnt 0" ::: "memory");

    // ---- reload P as A-fragment, accumulate ctx += P * V^T ----
    v16h aP = load_a16_lds(Pw, 40, lm, hf);
#pragma unroll
    for (int t4 = 0; t4 < 4; t4++) {
      v16h bV = load_b16_lds(Vts[buf] + t4 * 16 * 40, 40, lm, hf, 0);
      acc[t4] = WMMA_F16(aP, bV, acc[t4]);
    }
    __syncthreads();   // all waves done with this buffer before its next DMA
  }

  // ---- normalize and store ctx (f16, [b, s, head*64+d] row-major) ----
#pragma unroll
  for (int t4 = 0; t4 < 4; t4++) {
#pragma unroll
    for (int r = 0; r < 8; r++) {
      const int m = r + hf * 8;
      const float v = acc[t4][r] / lrow[r];
      ctxh[((size_t)batch * SEQ + qbase + m) * DIM + head * HD + t4 * 16 + lm] = (_Float16)v;
    }
  }
}

// ---------------------------------------------------------------------------
// Kernel 3: output projection ctx @ W_out^T + b_out -> f32 d_out
// (same LDS-staged, double-buffered A-tile scheme as qkv_gemm)
// ---------------------------------------------------------------------------
__global__ __launch_bounds__(128) void out_gemm(const _Float16* __restrict__ ctxh,
                                                const _Float16* __restrict__ wh,
                                                const float* __restrict__ bias,
                                                float* __restrict__ out) {
  __shared__ _Float16 At[2][64 * 40];
  const int tid  = threadIdx.x;
  const int lane = tid & 31, wid = tid >> 5;
  const int lm = lane & 15, hf = lane >> 4;
  const int batch = blockIdx.z;
  const int mbase = blockIdx.y * 64;
  const int nbase = (blockIdx.x * 4 + wid) * 32;

  const _Float16* Ag = ctxh + (size_t)batch * SEQ * DIM + (size_t)mbase * DIM;
  const _Float16* B0 = wh + (size_t)nbase * DIM;
  const _Float16* B1 = wh + (size_t)(nbase + 16) * DIM;

  issue_a_tile(Ag, At[0], 0, tid);
  v8f c[4][2] = {};
  int buf = 0;
  for (int k = 0; k < DIM; k += 32, buf ^= 1) {
    if (k + 32 < DIM) {
      issue_a_tile(Ag, At[buf ^ 1], k + 32, tid);
      S_WAIT_ASYNC2();
    } else {
      S_WAIT_ASYNC0();
    }
    __syncthreads();
    __builtin_prefetch(B0 + (size_t)lm * DIM + k + 64, 0, 1);
    v16h b0 = load_b16_g(B0 + k, DIM, lm, hf);
    v16h b1 = load_b16_g(B1 + k, DIM, lm, hf);
#pragma unroll
    for (int mi = 0; mi < 4; mi++) {
      v16h a = load_a16_lds(&At[buf][mi * 16 * 40], 40, lm, hf);
      c[mi][0] = WMMA_F16(a, b0, c[mi][0]);
      c[mi][1] = WMMA_F16(a, b1, c[mi][1]);
    }
    __syncthreads();
  }

#pragma unroll
  for (int ni = 0; ni < 2; ni++) {
    const int o = nbase + ni * 16 + lm;
    const float bv = bias[o];
#pragma unroll
    for (int mi = 0; mi < 4; mi++) {
#pragma unroll
      for (int r = 0; r < 8; r++) {
        const int srow = mbase + mi * 16 + r + hf * 8;
        out[((size_t)batch * SEQ + srow) * DIM + o] = c[mi][ni][r] + bv;
      }
    }
  }
}

// ---------------------------------------------------------------------------
// Launcher
// ---------------------------------------------------------------------------
extern "C" void kernel_launch(void* const* d_in, const int* in_sizes, int n_in,
                              void* d_out, int out_size, void* d_ws, size_t ws_size,
                              hipStream_t stream) {
  const float* x    = (const float*)d_in[0];  // [2, 2048, 1024]
  const float* Wkqv = (const float*)d_in[1];  // [3072, 1024]
  const float* bkqv = (const float*)d_in[2];  // [3072]
  const float* Wout = (const float*)d_in[3];  // [1024, 1024]
  const float* bout = (const float*)d_in[4];  // [1024]
  float* out = (float*)d_out;                 // [2, 2048, 1024]

  _Float16* ws = (_Float16*)d_ws;
  const size_t N_X    = (size_t)NBATCH * SEQ * DIM;       // 4194304
  const size_t N_WKQV = (size_t)3 * DIM * DIM;            // 3145728
  const size_t N_WOUT = (size_t)DIM * DIM;                // 1048576
  _Float16* xh    = ws;
  _Float16* wkqvh = xh + N_X;
  _Float16* wouth = wkqvh + N_WKQV;
  _Float16* Qh    = wouth + N_WOUT;
  _Float16* Kh    = Qh + N_X;
  _Float16* Vth   = Kh + N_X;
  _Float16* ctxh  = Vth + N_X;

  cvt_f32_f16<<<(int)((N_X / 4 + 255) / 256), 256, 0, stream>>>(x, xh, (int)(N_X / 4));
  cvt_f32_f16<<<(int)((N_WKQV / 4 + 255) / 256), 256, 0, stream>>>(Wkqv, wkqvh, (int)(N_WKQV / 4));
  cvt_f32_f16<<<(int)((N_WOUT / 4 + 255) / 256), 256, 0, stream>>>(Wout, wouth, (int)(N_WOUT / 4));

  // QKV: N = 3072 -> 96 wave tiles of 32 -> 24 blocks x 4 waves; M = 2048/64
  qkv_gemm<<<dim3(24, 32, NBATCH), 128, 0, stream>>>(xh, wkqvh, bkqv, Qh, Kh, Vth);

  // attention: 128 query tiles -> 32 blocks x 4 waves, per (head, batch)
  attn_flash<<<dim3(32, NHEAD, NBATCH), 128, 0, stream>>>(Qh, Kh, Vth, ctxh);

  // output projection: N = 1024 -> 32 wave tiles -> 8 blocks x 4 waves
  out_gemm<<<dim3(8, 32, NBATCH), 128, 0, stream>>>(ctxh, wouth, bout, out);
}
